// Transformer_levit_52853867544785
// MI455X (gfx1250) — matrix operand order repaired and checked by reading, not compiled, not run on hardware
//
#include <hip/hip_runtime.h>

typedef __attribute__((ext_vector_type(16))) _Float16 v16h;
typedef __attribute__((ext_vector_type(8)))  float    v8f;

#define EPSV 1e-5f

constexpr int BM = 128, BN = 64, BK = 32;
constexpr int MTOT = 32 * 1024;   // B * N rows

__device__ __forceinline__ v8f vzero8() {
  v8f z = {0.f, 0.f, 0.f, 0.f, 0.f, 0.f, 0.f, 0.f};
  return z;
}

// ---- DPP16 cross-lane moves (no LDS, no waits) ----------------------------
// quad_perm(1,0,3,2)=0xB1 (xor1), quad_perm(2,3,0,1)=0x4E (xor2),
// row_half_mirror=0x141 (xor7), row_mirror=0x140 (xor15)
template <int CTRL>
__device__ __forceinline__ float dppmov(float x) {
  return __int_as_float(
      __builtin_amdgcn_update_dpp(0, __float_as_int(x), CTRL, 0xF, 0xF, true));
}
__device__ __forceinline__ float red_max16(float x) {
  x = fmaxf(x, dppmov<0xB1>(x));
  x = fmaxf(x, dppmov<0x4E>(x));
  x = fmaxf(x, dppmov<0x141>(x));
  x = fmaxf(x, dppmov<0x140>(x));
  return x;
}
__device__ __forceinline__ float red_sum16(float x) {
  x += dppmov<0xB1>(x);
  x += dppmov<0x4E>(x);
  x += dppmov<0x141>(x);
  x += dppmov<0x140>(x);
  return x;
}

// ---- async global -> LDS (16B per lane), tracked by ASYNCcnt ---------------
__device__ __forceinline__ void async_b128(const void* gptr, void* lptr) {
  unsigned lds = (unsigned)(unsigned long long)lptr;  // addr[31:0] = LDS offset
  asm volatile("global_load_async_to_lds_b128 %0, %1, off"
               :: "v"(lds), "v"(gptr) : "memory");
}
template <int N>
__device__ __forceinline__ void wait_async() {
  asm volatile("s_wait_asynccnt %0" :: "i"(N) : "memory");
}

// ---------------------------------------------------------------------------
// f32 -> f16 convert (weights pre-pass; 320 KB total)
// ---------------------------------------------------------------------------
__global__ __launch_bounds__(256) void f32_to_f16_kernel(
    const float* __restrict__ src, _Float16* __restrict__ dst) {
  size_t i = (size_t)blockIdx.x * 256 + threadIdx.x;
  dst[i] = (_Float16)src[i];
}

// ---------------------------------------------------------------------------
// bias table: bt[h][n][m] = attn_biases[h][bias_idxs[n*1024+m]]  (f16, 8 MB)
// ---------------------------------------------------------------------------
__global__ __launch_bounds__(256) void bias_expand_kernel(
    const float* __restrict__ ab, const int* __restrict__ idxs,
    _Float16* __restrict__ bt) {
  size_t i = (size_t)blockIdx.x * 256 + threadIdx.x;   // 4*1024*1024 total
  int h  = (int)(i >> 20);
  int nm = (int)(i & 0xFFFFF);
  bt[i] = (_Float16)ab[h * 1024 + idxs[nm]];
}

// ---------------------------------------------------------------------------
// Fused GEMM + BatchNorm(+epilogue).  out[m][d] = bn(sum_k A[m][k]*Wh[d][k])
//  AMODE 0: A f32 row-major [M][KD]    (VALU convert-stage)
//  AMODE 1: A = x in (B,C,N) layout    (VALU convert-stage)
//  AMODE 2: A f16 row-major [M][KD]    (async-stage)
//  B tile always async-staged from pre-converted f16 weights.
//  EPI 0: QKV scatter (q*scale)  EPI 1: +x -> x1  EPI 2: hardtanh f16
//  EPI 3: +x1 -> (B,C,N) f32 output
// ---------------------------------------------------------------------------
template<int KD, int AMODE, int EPI, int ND>
__global__ __launch_bounds__(256) void gemm_bn_kernel(
    const float* __restrict__ A32, const _Float16* __restrict__ A16,
    const _Float16* __restrict__ Wh,
    const float* __restrict__ gg, const float* __restrict__ bb,
    const float* __restrict__ mmean, const float* __restrict__ vvar,
    const float* __restrict__ resid,
    float* __restrict__ outF, _Float16* __restrict__ outH,
    _Float16* __restrict__ qout, _Float16* __restrict__ kout,
    _Float16* __restrict__ vout) {
  __shared__ _Float16 As[BM][BK + 8];
  __shared__ _Float16 Bs[BN][BK + 8];   // [n][k]: fragment reads contiguous

  const int tid   = threadIdx.x;
  const int lane  = tid & 31;
  const int wv    = tid >> 5;
  const int mBase = blockIdx.x * BM;
  const int nBase = blockIdx.y * BN;

  v8f acc[4];
  #pragma unroll
  for (int t = 0; t < 4; ++t) acc[t] = vzero8();

  const int mrow  = wv * 16 + (lane & 15);
  const int kbA   = (lane >> 4) * 8;          // A frag K sub-offset
  const int ncol  = lane & 15;
  const int kOffB = (lane < 16) ? 0 : 16;     // B frag K offset

  for (int kt = 0; kt < KD / BK; ++kt) {
    const int k0 = kt * BK;
    __syncthreads();
    // ---- stage B tile (async, f16 weights) ----
    {
      int nn = tid >> 2;
      int kc = (tid & 3) * 8;
      async_b128(Wh + (size_t)(nBase + nn) * KD + k0 + kc, &Bs[nn][kc]);
    }
    // ---- stage A tile (BM x BK) ----
    if constexpr (AMODE == 1) {
      #pragma unroll
      for (int j = 0; j < (BM * BK) / 256; ++j) {
        int idx = j * 256 + tid;
        int mloc = idx & (BM - 1);
        int kk = idx >> 7;
        int m = mBase + mloc;
        int bI = m >> 10, n = m & 1023;
        As[mloc][kk] =
            (_Float16)A32[(size_t)bI * (KD * 1024) + (size_t)(k0 + kk) * 1024 + n];
      }
    } else if constexpr (AMODE == 0) {
      #pragma unroll
      for (int j = 0; j < (BM * BK) / 256; ++j) {
        int idx = j * 256 + tid;
        int kk = idx & (BK - 1);
        int mloc = idx >> 5;
        As[mloc][kk] = (_Float16)A32[(size_t)(mBase + mloc) * KD + k0 + kk];
      }
    } else {                                   // f16 A: async-stage
      #pragma unroll
      for (int j = 0; j < 2; ++j) {
        int c = j * 256 + tid;
        int mloc = c >> 2;
        int kc = (c & 3) * 8;
        async_b128(A16 + (size_t)(mBase + mloc) * KD + k0 + kc, &As[mloc][kc]);
      }
    }
    wait_async<0>();
    __syncthreads();

    // ---- WMMA: one A frag (16x32), 4 B frags (32x16) ----
    v16h a;
    #pragma unroll
    for (int h = 0; h < 8; ++h) a[h] = As[mrow][kbA + h];
    #pragma unroll
    for (int h = 0; h < 8; ++h) a[8 + h] = As[mrow][16 + kbA + h];

    #pragma unroll
    for (int t = 0; t < 4; ++t) {
      v16h bf;
      #pragma unroll
      for (int h = 0; h < 16; ++h) bf[h] = Bs[t * 16 + ncol][(h & 8) * 2 + kbA + (h & 7)];
      acc[t] = __builtin_amdgcn_wmma_f32_16x16x32_f16(
          false, a, false, bf, (short)0, acc[t], false, false);
    }
  }

  // ---- epilogue: BN fold + per-mode routing ----
  const int half8 = (lane >> 4) * 8;
  #pragma unroll
  for (int t = 0; t < 4; ++t) {
    const int d = nBase + t * 16 + ncol;
    const float sc = gg[d] * rsqrtf(vvar[d] + EPSV);
    const float bi = bb[d] - mmean[d] * sc;
    #pragma unroll
    for (int r = 0; r < 8; ++r) {
      const int m = mBase + wv * 16 + half8 + r;
      float val = acc[t][r] * sc + bi;
      if constexpr (EPI == 0) {                 // QKV scatter
        int bI = m >> 10, n = m & 1023;
        int head = d >> 7, rem = d & 127;
        size_t row = (size_t)(bI * 4 + head) * 1024 + n;
        if (rem < 32)
          qout[row * 32 + rem] = (_Float16)(val * 0.17677669529663687f);
        else if (rem < 64)
          kout[row * 32 + (rem - 32)] = (_Float16)val;
        else
          vout[row * 64 + (rem - 64)] = (_Float16)val;
      } else if constexpr (EPI == 1) {          // proj: + x -> x1
        int bI = m >> 10, n = m & 1023;
        outF[(size_t)m * ND + d] =
            val + resid[(size_t)bI * (128 * 1024) + (size_t)d * 1024 + n];
      } else if constexpr (EPI == 2) {          // f1: hardtanh -> h1 f16
        outH[(size_t)m * ND + d] = (_Float16)fminf(fmaxf(val, -1.0f), 1.0f);
      } else {                                  // f2: + x1, transposed store
        int bI = m >> 10, n = m & 1023;
        outF[(size_t)bI * (128 * 1024) + (size_t)d * 1024 + n] =
            val + resid[(size_t)m * 128 + d];
      }
    }
  }
}

// ---------------------------------------------------------------------------
// Flash attention: grid (8 query-blocks, B*H).  Each wave owns 16 query rows.
// K/V tiles double-buffered with GLOBAL_LOAD_ASYNC_TO_LDS_B128: exactly 3
// async instructions per block per wave, so `s_wait_asynccnt 3` throttles to
// "previous block landed" while the next block's DMA overlaps compute.
// ---------------------------------------------------------------------------
__global__ __launch_bounds__(256) void attn_kernel(
    const _Float16* __restrict__ q, const _Float16* __restrict__ kg,
    const _Float16* __restrict__ vg, const _Float16* __restrict__ bt,
    _Float16* __restrict__ o) {
  __shared__ _Float16 Ks[2][64][32];
  __shared__ _Float16 Vs[2][64][64];
  __shared__ _Float16 Ps[8][16][68];

  const int tid  = threadIdx.x;
  const int lane = tid & 31;
  const int wv   = tid >> 5;
  const int bh   = blockIdx.y;        // b*4 + h
  const int hh   = bh & 3;
  const int bI   = bh >> 2;
  const int qw   = blockIdx.x * 128 + wv * 16;

  const size_t qkBase = (size_t)bh * 1024;
  const int mrow  = lane & 15;
  const int kbA   = (lane >> 4) * 8;
  const int ncol  = lane & 15;
  const int kOffB = (lane < 16) ? 0 : 16;
  const int half8 = (lane >> 4) * 8;

  // per-thread async chunk coordinates (16 B = 8 halves each)
  const int kR = tid >> 2, kC = (tid & 3) * 8;     // K: 64x32, 1 chunk/thread

  auto issue = [&](int blk, int buf) {
    async_b128(kg + (qkBase + blk * 64 + kR) * 32 + kC, &Ks[buf][kR][kC]);
    #pragma unroll
    for (int j = 0; j < 2; ++j) {
      int c = j * 256 + tid;
      int vR = c >> 3, vC = (c & 7) * 8;
      async_b128(vg + (qkBase + blk * 64 + vR) * 64 + vC, &Vs[buf][vR][vC]);
    }
  };

  // Q fragment held in registers for the whole pass (16x32, scale pre-applied)
  v16h aq;
  {
    const _Float16* qrow = q + (qkBase + qw + mrow) * 32;
    #pragma unroll
    for (int h = 0; h < 8; ++h) aq[h] = qrow[kbA + h];
    #pragma unroll
    for (int h = 0; h < 8; ++h) aq[8 + h] = qrow[16 + kbA + h];
  }

  v8f acc[4];
  #pragma unroll
  for (int t = 0; t < 4; ++t) acc[t] = vzero8();
  float mst[8], lst[8];
  #pragma unroll
  for (int r = 0; r < 8; ++r) { mst[r] = -1e30f; lst[r] = 0.0f; }

  issue(0, 0);                          // prologue DMA for block 0

  for (int kb = 0; kb < 16; ++kb) {
    const int buf = kb & 1;
    if (kb + 1 < 16) {
      issue(kb + 1, buf ^ 1);           // overlaps compute of block kb
      wait_async<3>();                  // block kb landed (in-order completion)
    } else {
      wait_async<0>();
    }
    __syncthreads();                    // all waves' block-kb tiles visible

    // S = Q * K^T  (four 16x16 tiles over 64 keys) + bias
    v8f s[4];
    #pragma unroll
    for (int t = 0; t < 4; ++t) {
      v16h bk;
      #pragma unroll
      for (int h = 0; h < 16; ++h) bk[h] = Ks[buf][t * 16 + ncol][(h & 8) * 2 + kbA + (h & 7)];
      s[t] = __builtin_amdgcn_wmma_f32_16x16x32_f16(
          false, aq, false, bk, (short)0, vzero8(), false, false);
    }
    #pragma unroll
    for (int t = 0; t < 4; ++t) {
      #pragma unroll
      for (int r = 0; r < 8; ++r) {
        int qn = qw + half8 + r;
        int kc = kb * 64 + t * 16 + ncol;
        s[t][r] += (float)bt[((size_t)hh << 20) + ((size_t)qn << 10) + kc];
      }
    }

    // online softmax (row lives in 16 lanes of one half-wave; DPP reductions)
    #pragma unroll
    for (int r = 0; r < 8; ++r) {
      float mx = fmaxf(fmaxf(s[0][r], s[1][r]), fmaxf(s[2][r], s[3][r]));
      mx = red_max16(mx);
      float mnew  = fmaxf(mst[r], mx);
      float alpha = __expf(mst[r] - mnew);
      float rs = 0.0f;
      #pragma unroll
      for (int t = 0; t < 4; ++t) {
        float p = __expf(s[t][r] - mnew);
        s[t][r] = p;
        rs += p;
      }
      rs = red_sum16(rs);
      lst[r] = lst[r] * alpha + rs;
      mst[r] = mnew;
      #pragma unroll
      for (int t = 0; t < 4; ++t) acc[t][r] *= alpha;
    }

    // P -> per-wave LDS buffer (f16), C-layout -> row-major 16x64
    #pragma unroll
    for (int t = 0; t < 4; ++t) {
      #pragma unroll
      for (int r = 0; r < 8; ++r)
        Ps[wv][half8 + r][t * 16 + ncol] = (_Float16)s[t][r];
    }

    // O += P * V   (K = 64 keys -> two WMMA K-steps, 4 vdim tiles)
    #pragma unroll
    for (int kk = 0; kk < 2; ++kk) {
      v16h ap;
      const _Float16* pr = &Ps[wv][mrow][kk * 32];
      #pragma unroll
      for (int h = 0; h < 8; ++h) ap[h] = pr[kbA + h];
      #pragma unroll
      for (int h = 0; h < 8; ++h) ap[8 + h] = pr[16 + kbA + h];
      #pragma unroll
      for (int t = 0; t < 4; ++t) {
        v16h bv;
        #pragma unroll
        for (int h = 0; h < 16; ++h)
          bv[h] = Vs[buf][kk * 32 + (h & 8) * 2 + kbA + (h & 7)][t * 16 + ncol];
        acc[t] = __builtin_amdgcn_wmma_f32_16x16x32_f16(
            false, ap, false, bv, (short)0, acc[t], false, false);
      }
    }
    __syncthreads();                    // block-kb reads done: buffer reusable
  }

  // normalize, hardtanh, store o[B][N][256] f16 (ready as proj GEMM A)
  #pragma unroll
  for (int r = 0; r < 8; ++r) {
    float inv = 1.0f / lst[r];
    int qn = qw + half8 + r;
    size_t orow = ((size_t)bI * 1024 + qn) * 256 + hh * 64;
    #pragma unroll
    for (int t = 0; t < 4; ++t) {
      float val = fminf(fmaxf(acc[t][r] * inv, -1.0f), 1.0f);
      o[orow + t * 16 + ncol] = (_Float16)val;
    }
  }
}

// ---------------------------------------------------------------------------
extern "C" void kernel_launch(void* const* d_in, const int* in_sizes, int n_in,
                              void* d_out, int out_size, void* d_ws,
                              size_t ws_size, hipStream_t stream) {
  (void)in_sizes; (void)n_in; (void)out_size; (void)ws_size;
  const float* x       = (const float*)d_in[0];
  const float* W_qkv   = (const float*)d_in[1];
  const float* g_qkv   = (const float*)d_in[2];
  const float* b_qkv   = (const float*)d_in[3];
  const float* m_qkv   = (const float*)d_in[4];
  const float* v_qkv   = (const float*)d_in[5];
  const float* W_proj  = (const float*)d_in[6];
  const float* g_proj  = (const float*)d_in[7];
  const float* b_proj  = (const float*)d_in[8];
  const float* m_proj  = (const float*)d_in[9];
  const float* v_proj  = (const float*)d_in[10];
  const float* W_f1    = (const float*)d_in[11];
  const float* g_f1    = (const float*)d_in[12];
  const float* b_f1    = (const float*)d_in[13];
  const float* m_f1    = (const float*)d_in[14];
  const float* v_f1    = (const float*)d_in[15];
  const float* W_f2    = (const float*)d_in[16];
  const float* g_f2    = (const float*)d_in[17];
  const float* b_f2    = (const float*)d_in[18];
  const float* m_f2    = (const float*)d_in[19];
  const float* v_f2    = (const float*)d_in[20];
  const float* attn_b  = (const float*)d_in[21];
  const int*   b_idxs  = (const int*)d_in[22];
  float* out = (float*)d_out;

  // workspace layout (halves); x1 aliases q+k (dead after attn), h1 aliases o
  _Float16* qb    = (_Float16*)d_ws;                   //  8 MB
  _Float16* kb    = qb + (size_t)4 * 1024 * 1024;      //  8 MB
  _Float16* vb    = kb + (size_t)4 * 1024 * 1024;      // 16 MB
  _Float16* bt    = vb + (size_t)8 * 1024 * 1024;      //  8 MB
  _Float16* ob    = bt + (size_t)4 * 1024 * 1024;      // 16 MB
  _Float16* whq   = ob + (size_t)8 * 1024 * 1024;      // 128 KB (512x128 f16)
  _Float16* whp   = whq + 512 * 128;                   //  64 KB (128x256 f16)
  _Float16* whf1  = whp + 128 * 256;                   //  64 KB (256x128 f16)
  _Float16* whf2  = whf1 + 256 * 128;                  //  64 KB (128x256 f16)
  float*    x1    = (float*)d_ws;                      // 16 MB (over q+k)
  _Float16* h1    = ob;                                // 16 MB (over o)

  // weight f32 -> f16 pre-pass (tiny)
  f32_to_f16_kernel<<<(512 * 128) / 256, 256, 0, stream>>>(W_qkv, whq);
  f32_to_f16_kernel<<<(128 * 256) / 256, 256, 0, stream>>>(W_proj, whp);
  f32_to_f16_kernel<<<(256 * 128) / 256, 256, 0, stream>>>(W_f1, whf1);
  f32_to_f16_kernel<<<(128 * 256) / 256, 256, 0, stream>>>(W_f2, whf2);

  bias_expand_kernel<<<16384, 256, 0, stream>>>(attn_b, b_idxs, bt);

  gemm_bn_kernel<128, 1, 0, 512>
      <<<dim3(MTOT / BM, 512 / BN), 256, 0, stream>>>(
          x, nullptr, whq, g_qkv, b_qkv, m_qkv, v_qkv,
          nullptr, nullptr, nullptr, qb, kb, vb);

  attn_kernel<<<dim3(8, 128), 256, 0, stream>>>(qb, kb, vb, bt, ob);

  gemm_bn_kernel<256, 2, 1, 128>
      <<<dim3(MTOT / BM, 128 / BN), 256, 0, stream>>>(
          nullptr, ob, whp, g_proj, b_proj, m_proj, v_proj,
          x, x1, nullptr, nullptr, nullptr, nullptr);

  gemm_bn_kernel<128, 0, 2, 256>
      <<<dim3(MTOT / BM, 256 / BN), 256, 0, stream>>>(
          x1, nullptr, whf1, g_f1, b_f1, m_f1, v_f1,
          nullptr, nullptr, h1, nullptr, nullptr, nullptr);

  gemm_bn_kernel<256, 2, 3, 128>
      <<<dim3(MTOT / BM, 128 / BN), 256, 0, stream>>>(
          nullptr, h1, whf2, g_f2, b_f2, m_f2, v_f2,
          x1, out, nullptr, nullptr, nullptr, nullptr);
}
